// SelfAttention_21457656611341
// MI455X (gfx1250) — compile-verified, matrix operand order checked by reading, not run on hardware
//
#include <hip/hip_runtime.h>

// Problem constants (match reference): B=4, C=256, H=W=64 -> N=4096, CQK=32
#define Bn  4
#define Cc  256
#define Nn  4096
#define DQK 32

typedef unsigned short u16;
typedef __attribute__((ext_vector_type(16))) __bf16 v16bf;
typedef __attribute__((ext_vector_type(8)))  float  v8f;
typedef __attribute__((ext_vector_type(4)))  unsigned int u32x4;
typedef __attribute__((ext_vector_type(4)))  int v4i;

// Async global->LDS path (CDNA5 ASYNCcnt) if the toolchain exposes it.
#if defined(__has_builtin)
#  if __has_builtin(__builtin_amdgcn_global_load_async_to_lds_b128) && \
      __has_builtin(__builtin_amdgcn_s_wait_asynccnt)
#    define HAVE_ASYNC_LDS 1
#  endif
#endif
#ifndef HAVE_ASYNC_LDS
#  define HAVE_ASYNC_LDS 0
#endif

union TileU {
    v16bf v;
    u32x4 q[2];
    unsigned u[8];
};

// round-to-nearest-even f32 -> bf16 (bit trick)
__device__ __forceinline__ u16 f2bf(float f) {
    unsigned u = __float_as_uint(f);
    u = (u + 0x7FFFu + ((u >> 16) & 1u)) >> 16;
    return (u16)u;
}

__device__ __forceinline__ unsigned pack2bf(float a, float b) {
    unsigned ua = __float_as_uint(a), ub = __float_as_uint(b);
    ua = (ua + 0x7FFFu + ((ua >> 16) & 1u)) >> 16;
    ub = (ub + 0x7FFFu + ((ub >> 16) & 1u)) >> 16;
    return ua | (ub << 16);
}

// Load one WMMA 16-bit operand tile (A or B) from a [pos][32] bf16 array.
// 16-bit A 16x32 layout: lanes 0-15 hold row m, K=0..7 (VGPR0-3) and 16..23
// (VGPR4-7); lanes 16-31 hold row m, K=8..15 and 24..31.  The B (32x16)
// operand has the identical per-lane pattern with n in the role of m.
// 'row' must already include (lane & 15).
__device__ __forceinline__ v16bf load_posrow(const u16* __restrict__ base, int row, int lane) {
    const u16* p = base + row * DQK + ((lane & 16) ? 8 : 0);
    TileU t;
    t.q[0] = *(const u32x4*)(p);
    t.q[1] = *(const u32x4*)(p + 16);
    return t.v;
}

#if HAVE_ASYNC_LDS
// Stage 64 B (one 32-n bf16 channel row) global -> LDS via the async engine.
__device__ __forceinline__ void stage_async64(const u16* gsrc, u16* ldst) {
#pragma unroll
    for (int j = 0; j < 4; ++j) {
        __builtin_amdgcn_global_load_async_to_lds_b128(
            (__attribute__((address_space(1))) v4i*)(gsrc + 8 * j),
            (__attribute__((address_space(3))) v4i*)(ldst + 8 * j),
            0, 0);
    }
}
#endif

// ---------------------------------------------------------------------------
// Kernel 1: 1x1-conv projections.  qT,kT stored bf16 [B][N][32] (position-major,
// WMMA-operand friendly), v stored bf16 [B][C][N].
// Register-blocked 4 output channels/thread: one LDS read feeds 4 FMAs; the
// four W rows are wave-uniform -> scalar loads.
// Grid: (N/32, B), 256 threads. LDS tile of x: [256 c][32 n] fp32 = 32 KB.
// ---------------------------------------------------------------------------
__global__ __launch_bounds__(256) void proj_kernel(
    const float* __restrict__ x,
    const float* __restrict__ Wq, const float* __restrict__ bq,
    const float* __restrict__ Wk, const float* __restrict__ bk,
    const float* __restrict__ Wv, const float* __restrict__ bv,
    u16* __restrict__ qT, u16* __restrict__ kT, u16* __restrict__ vb)
{
    __shared__ float xs[Cc * 32];
    const int b  = blockIdx.y;
    const int n0 = blockIdx.x * 32;
    const int t  = threadIdx.x;
    const int g  = t >> 5;     // wave id 0..7 (uniform per wave)
    const int n  = t & 31;     // position within tile

    const float* xb = x + ((size_t)b * Cc) * Nn + n0;
#pragma unroll
    for (int k = 0; k < 32; ++k) {
        int c = g + 8 * k;
        xs[c * 32 + n] = xb[(size_t)c * Nn + n];
    }
    __syncthreads();

    // 320 output rows over 8 waves, 4-wide register blocking.  Group bases are
    // multiples of 4 and the q/k/v boundaries (32, 64) are too, so a group
    // never straddles a projection boundary.
    for (int i = 0; i < 10; ++i) {
        int o = g * 40 + i * 4;
        const float* Wr;
        const float* br;
        int kind;
        if (o < 32)      { Wr = Wq + o * Cc;        br = bq + o;        kind = 0; }
        else if (o < 64) { Wr = Wk + (o - 32) * Cc; br = bk + (o - 32); kind = 1; }
        else             { Wr = Wv + (o - 64) * Cc; br = bv + (o - 64); kind = 2; }
        float a0 = br[0], a1 = br[1], a2 = br[2], a3 = br[3];
#pragma unroll 4
        for (int c = 0; c < Cc; ++c) {
            float xv = xs[c * 32 + n];
            a0 += Wr[c] * xv;
            a1 += Wr[Cc + c] * xv;
            a2 += Wr[2 * Cc + c] * xv;
            a3 += Wr[3 * Cc + c] * xv;
        }
        if (kind == 0) {        // 4 consecutive channels at one position: 8-B store
            unsigned long long hh = (unsigned long long)pack2bf(a0, a1)
                                  | ((unsigned long long)pack2bf(a2, a3) << 32);
            *(unsigned long long*)(qT + ((size_t)b * Nn + n0 + n) * DQK + o) = hh;
        } else if (kind == 1) {
            unsigned long long hh = (unsigned long long)pack2bf(a0, a1)
                                  | ((unsigned long long)pack2bf(a2, a3) << 32);
            *(unsigned long long*)(kT + ((size_t)b * Nn + n0 + n) * DQK + (o - 32)) = hh;
        } else {                // v rows: stride N apart
            size_t base = ((size_t)b * Cc + (o - 64)) * Nn + n0 + n;
            vb[base]          = f2bf(a0);
            vb[base + Nn]     = f2bf(a1);
            vb[base + 2 * Nn] = f2bf(a2);
            vb[base + 3 * Nn] = f2bf(a3);
        }
    }
}

// ---------------------------------------------------------------------------
// Kernel 2: softmax statistics over the contracted axis n for each column m.
// scores[n,m] = q[:,n]·k[:,m] via one v_wmma_f32_16x16x32_bf16 per 16 n's.
// Streaming logsumexp; cross-lane-half and cross-wave merge through LDS.
// Grid: (N/16, B), 128 threads (4 waves).
// ---------------------------------------------------------------------------
__global__ __launch_bounds__(128) void stats_kernel(
    const u16* __restrict__ qT, const u16* __restrict__ kT,
    float* __restrict__ rowmax, float* __restrict__ rowsum)
{
    __shared__ float smax[128];
    __shared__ float ssum[128];
    const int b    = blockIdx.y;
    const int m0   = blockIdx.x * 16;
    const int t    = threadIdx.x;
    const int lane = t & 31;
    const int wave = t >> 5;
    const int lm   = lane & 15;

    const u16* qb = qT + (size_t)b * Nn * DQK;
    const u16* kb = kT + (size_t)b * Nn * DQK;

    const v16bf Bk = load_posrow(kb, m0 + lm, lane);  // B operand fixed: columns m

    float rmax = -3.0e38f, rsum = 0.0f;
    for (int n0 = wave * 16; n0 < Nn; n0 += 64) {
        __builtin_prefetch(qb + (size_t)(n0 + 64 + lm) * DQK, 0, 3);  // near-scope
        v16bf Aq = load_posrow(qb, n0 + lm, lane);    // A operand: rows n
        v8f z = {0.f, 0.f, 0.f, 0.f, 0.f, 0.f, 0.f, 0.f};
        v8f s = __builtin_amdgcn_wmma_f32_16x16x32_bf16(
            false, Aq, false, Bk, (short)0, z, false, false);
        float tmax = s[0];
#pragma unroll
        for (int r = 1; r < 8; ++r) tmax = fmaxf(tmax, s[r]);
        float nm = fmaxf(rmax, tmax);
        float es = 0.0f;
#pragma unroll
        for (int r = 0; r < 8; ++r) es += __expf(s[r] - nm);
        rsum = rsum * __expf(rmax - nm) + es;
        rmax = nm;
    }
    smax[t] = rmax;
    ssum[t] = rsum;
    __syncthreads();
    if (t < 16) {   // merge 4 waves x 2 lane-halves for column m = m0 + t
        float M = -3.0e38f, S = 0.0f;
#pragma unroll
        for (int w = 0; w < 4; ++w)
#pragma unroll
            for (int h = 0; h < 2; ++h) {
                float m2 = smax[w * 32 + h * 16 + t];
                float s2 = ssum[w * 32 + h * 16 + t];
                float nm = fmaxf(M, m2);
                S = S * __expf(M - nm) + s2 * __expf(m2 - nm);
                M = nm;
            }
        rowmax[(size_t)b * Nn + m0 + t] = M;
        rowsum[(size_t)b * Nn + m0 + t] = S;
    }
}

// ---------------------------------------------------------------------------
// Kernel 3: fused attention output.  Each wave owns 16 output columns m and all
// 256 channels (16 c-tiles of v8f fp32 accumulators = 128 VGPRs).  Per 32-n
// chunk: v[256c x 32n] double-buffered in LDS (80-B row pitch -> conflict-free
// ds_load_b128), 2 score WMMAs, exp+pack into the bf16 B-operand layout,
// 16 PV WMMAs.  The v-chunk copy uses the CDNA5 async global->LDS engine
// (ASYNCcnt) when available, else a register-pipelined sync copy; either way
// chunk i+1 streams in while chunk i's 18 WMMAs execute, one barrier/chunk.
// Grid: (N/128, B), 256 threads (8 waves).
// ---------------------------------------------------------------------------
__global__ __launch_bounds__(256) void attn_out_kernel(
    const float* __restrict__ x,
    const u16* __restrict__ qT, const u16* __restrict__ kT, const u16* __restrict__ vb,
    const float* __restrict__ rowmax, const float* __restrict__ rowsum,
    const float* __restrict__ gamma, float* __restrict__ out)
{
    __shared__ __align__(16) u16 vs[2 * Cc * 40];   // 2 x (256 rows x 80 B pitch)
    const int b    = blockIdx.y;
    const int t    = threadIdx.x;
    const int lane = t & 31;
    const int wave = t >> 5;
    const int lm   = lane & 15;
    const int m0   = blockIdx.x * 128 + wave * 16;
    const int koff = (lane & 16) ? 8 : 0;

    const u16* qb = qT + (size_t)b * Nn * DQK;
    const u16* kb = kT + (size_t)b * Nn * DQK;

    const v16bf Bk  = load_posrow(kb, m0 + lm, lane);
    const float rmx = rowmax[(size_t)b * Nn + m0 + lm];
    const float rsi = 1.0f / rowsum[(size_t)b * Nn + m0 + lm];

    v8f acc[16];
#pragma unroll
    for (int ct = 0; ct < 16; ++ct)
        acc[ct] = (v8f){0.f, 0.f, 0.f, 0.f, 0.f, 0.f, 0.f, 0.f};

    const u16* vrow = vb + ((size_t)b * Cc + t) * Nn;   // thread t stages channel c=t

#if HAVE_ASYNC_LDS
    stage_async64(vrow, vs + t * 40);                    // chunk 0 -> buffer 0
#else
    u32x4 ra, rb2, rc, rd;
    {
        const u32x4* s = (const u32x4*)(vrow);
        ra = s[0]; rb2 = s[1]; rc = s[2]; rd = s[3];
    }
#endif

    int it = 0;
    for (int n0 = 0; n0 < Nn; n0 += 32, ++it) {
        const int cur = it & 1;
        u16* buf = vs + cur * (Cc * 40);

#if HAVE_ASYNC_LDS
        __builtin_amdgcn_s_wait_asynccnt(0);   // own chunk-i copies landed
        __syncthreads();                       // everyone's chunk-i copies landed
        if (n0 + 32 < Nn)                      // overlap chunk i+1 with compute
            stage_async64(vrow + n0 + 32, vs + (cur ^ 1) * (Cc * 40) + t * 40);
#else
        {
            u32x4* dst = (u32x4*)buf + t * 5;  // 80-B row pitch
            dst[0] = ra; dst[1] = rb2; dst[2] = rc; dst[3] = rd;
        }
        if (n0 + 32 < Nn) {
            const u32x4* s = (const u32x4*)(vrow + n0 + 32);
            ra = s[0]; rb2 = s[1]; rc = s[2]; rd = s[3];
        }
        __syncthreads();
#endif
        // speculative near-scope prefetch of the next chunk's q rows
        __builtin_prefetch(qb + (size_t)(n0 + 32 + lm) * DQK, 0, 3);

        v16bf Aq0 = load_posrow(qb, n0 + lm, lane);
        v16bf Aq1 = load_posrow(qb, n0 + 16 + lm, lane);
        v8f z = {0.f, 0.f, 0.f, 0.f, 0.f, 0.f, 0.f, 0.f};
        v8f s0 = __builtin_amdgcn_wmma_f32_16x16x32_bf16(
            false, Aq0, false, Bk, (short)0, z, false, false);
        v8f s1 = __builtin_amdgcn_wmma_f32_16x16x32_bf16(
            false, Aq1, false, Bk, (short)0, z, false, false);

        // exp tile -> bf16 B-operand (D-tile rows map 1:1 onto B K-packing)
        TileU P;
#pragma unroll
        for (int gq = 0; gq < 4; ++gq) {
            P.u[gq]     = pack2bf(__expf(s0[2 * gq] - rmx), __expf(s0[2 * gq + 1] - rmx));
            P.u[4 + gq] = pack2bf(__expf(s1[2 * gq] - rmx), __expf(s1[2 * gq + 1] - rmx));
        }

#pragma unroll
        for (int ct = 0; ct < 16; ++ct) {
            const u16* p = buf + (ct * 16 + lm) * 40 + koff;
            TileU Av;
            Av.q[0] = *(const u32x4*)(p);
            Av.q[1] = *(const u32x4*)(p + 16);
            acc[ct] = __builtin_amdgcn_wmma_f32_16x16x32_bf16(
                false, Av.v, false, P.v, (short)0, acc[ct], false, false);
        }
    }

    const float g = gamma[0];
    const int crow = (lane & 16) ? 8 : 0;
#pragma unroll
    for (int ct = 0; ct < 16; ++ct) {
#pragma unroll
        for (int r = 0; r < 8; ++r) {
            int c = ct * 16 + r + crow;
            size_t idx = ((size_t)b * Cc + c) * (size_t)Nn + m0 + lm;
            out[idx] = g * acc[ct][r] * rsi + x[idx];
        }
    }
}

// ---------------------------------------------------------------------------
extern "C" void kernel_launch(void* const* d_in, const int* in_sizes, int n_in,
                              void* d_out, int out_size, void* d_ws, size_t ws_size,
                              hipStream_t stream)
{
    const float* x     = (const float*)d_in[0];
    const float* Wq    = (const float*)d_in[1];
    const float* bq    = (const float*)d_in[2];
    const float* Wk    = (const float*)d_in[3];
    const float* bk    = (const float*)d_in[4];
    const float* Wv    = (const float*)d_in[5];
    const float* bv    = (const float*)d_in[6];
    const float* gamma = (const float*)d_in[7];
    float* out = (float*)d_out;

    char* ws = (char*)d_ws;
    u16*   qT     = (u16*)(ws);                              // 1 MB
    u16*   kT     = (u16*)(ws + (1u << 20));                 // 1 MB
    u16*   vb     = (u16*)(ws + (2u << 20));                 // 8 MB
    float* rowmax = (float*)(ws + (10u << 20));              // 64 KB
    float* rowsum = (float*)(ws + (10u << 20) + Bn * Nn * sizeof(float));

    proj_kernel<<<dim3(Nn / 32, Bn), 256, 0, stream>>>(
        x, Wq, bq, Wk, bk, Wv, bv, qT, kT, vb);
    stats_kernel<<<dim3(Nn / 16, Bn), 128, 0, stream>>>(
        qT, kT, rowmax, rowsum);
    attn_out_kernel<<<dim3(Nn / 128, Bn), 256, 0, stream>>>(
        x, qT, kT, vb, rowmax, rowsum, gamma, out);
}
